// GPTModel_87935160418703
// MI455X (gfx1250) — matrix operand order repaired and checked by reading, not compiled, not run on hardware
//
#include <hip/hip_runtime.h>
#include <hip/hip_bf16.h>

// ---- model constants ------------------------------------------------------
#define BB   2
#define TT   2048
#define DD   512
#define LL   4
#define HH   8
#define FFF  2048
#define VV   32000
#define DHH  64
#define MM   (BB*TT)            // 4096 rows of activations

typedef __attribute__((ext_vector_type(16))) __bf16 v16bf;
typedef __attribute__((ext_vector_type(8)))  float  v8f;
typedef __attribute__((ext_vector_type(4)))  int    i32x4;

union AF { v16bf v; uint4 q[2]; };
union QU { uint4 u; __bf16 e[8]; };

// CDNA5 async global->LDS path (ASYNCcnt), guarded so either toolchain compiles.
#if defined(__gfx1250__) && __has_builtin(__builtin_amdgcn_global_load_async_to_lds_b128)
#define USE_ASYNC 1
#else
#define USE_ASYNC 0
#endif

__device__ __forceinline__ void async_b128(const void* g, void* l) {
#if USE_ASYNC
  __builtin_amdgcn_global_load_async_to_lds_b128(
      (__attribute__((address_space(1))) i32x4*)g,
      (__attribute__((address_space(3))) i32x4*)l, 0, 0);
#endif
}

__device__ __forceinline__ void wait_async_zero() {
#if USE_ASYNC
#if __has_builtin(__builtin_amdgcn_s_wait_asynccnt)
  __builtin_amdgcn_s_wait_asynccnt(0);
#else
  asm volatile("s_wait_asynccnt 0x0" ::: "memory");
#endif
#endif
}

__device__ __forceinline__ v8f vzero8() {
  v8f z;
#pragma unroll
  for (int i = 0; i < 8; ++i) z[i] = 0.0f;
  return z;
}

__device__ __forceinline__ v8f wmma_bf16(v16bf a, v16bf b, v8f c) {
  return __builtin_amdgcn_wmma_f32_16x16x32_bf16(false, a, false, b,
                                                 (short)0, c, false, false);
}

// ---- elementwise helpers --------------------------------------------------
__global__ void cvt_kernel(const float* __restrict__ in, __bf16* __restrict__ out, int n) {
  int i = blockIdx.x * 256 + threadIdx.x;
  if (i < n) out[i] = (__bf16)in[i];
}

__global__ void embed_kernel(const int* __restrict__ ids,
                             const float* __restrict__ tok,
                             const float* __restrict__ pos,
                             float* __restrict__ X) {
  int m = blockIdx.x;                 // 0..4095
  int t = m & (TT - 1);
  int id = ids[m];
  int i = threadIdx.x;
  X[(size_t)m * DD + i]       = tok[(size_t)id * DD + i]       + pos[(size_t)t * DD + i];
  X[(size_t)m * DD + i + 256] = tok[(size_t)id * DD + i + 256] + pos[(size_t)t * DD + i + 256];
}

__global__ void ln_kernel(const float* __restrict__ X,
                          const float* __restrict__ w,
                          const float* __restrict__ b,
                          __bf16* __restrict__ out) {
  int row = blockIdx.x, tid = threadIdx.x;
  const float* xr = X + (size_t)row * DD;
  float a0 = xr[tid], a1 = xr[tid + 256];
  __shared__ float s1[256], s2[256];
  s1[tid] = a0 + a1;
  s2[tid] = a0 * a0 + a1 * a1;
  __syncthreads();
  for (int off = 128; off > 0; off >>= 1) {
    if (tid < off) { s1[tid] += s1[tid + off]; s2[tid] += s2[tid + off]; }
    __syncthreads();
  }
  float mu  = s1[0] * (1.0f / DD);
  float var = s2[0] * (1.0f / DD) - mu * mu;
  float rs  = rsqrtf(var + 1e-5f);
  out[(size_t)row * DD + tid]       = (__bf16)((a0 - mu) * rs * w[tid]       + b[tid]);
  out[(size_t)row * DD + tid + 256] = (__bf16)((a1 - mu) * rs * w[tid + 256] + b[tid + 256]);
}

// qkv fp32 [M, 3D] -> Q/K/V bf16 [B,H,T,DH]
__global__ void qkv_split_kernel(const float* __restrict__ QKV,
                                 __bf16* __restrict__ Qb,
                                 __bf16* __restrict__ Kb,
                                 __bf16* __restrict__ Vb) {
  int m = blockIdx.x;
  int b = m >> 11, t = m & (TT - 1);
  for (int c = threadIdx.x; c < 3 * DD; c += 256) {
    float v = QKV[(size_t)m * (3 * DD) + c];
    int which = c / DD;
    int rem   = c - which * DD;
    int h  = rem >> 6;
    int dh = rem & 63;
    size_t di = (((size_t)(b * HH + h)) * TT + t) * DHH + dh;
    __bf16 bv = (__bf16)v;
    if      (which == 0) Qb[di] = bv;
    else if (which == 1) Kb[di] = bv;
    else                 Vb[di] = bv;
  }
}

// ---- WMMA GEMM: C[M,N] = A[M,K] * W[N,K]^T  (A,W bf16, acc f32) -----------
// Double-buffered LDS pipeline; async global->LDS when available.
// MODE 0: OF = acc        MODE 1: OF += acc        MODE 2: OB = bf16(gelu(acc))
template <int MODE>
__global__ __launch_bounds__(256) void gemm_kernel(const __bf16* __restrict__ A,
                                                   const __bf16* __restrict__ W,
                                                   float* __restrict__ OF,
                                                   __bf16* __restrict__ OB,
                                                   int K, int N) {
  __shared__ __attribute__((aligned(16))) __bf16 As[2][128 * 32];
  __shared__ __attribute__((aligned(16))) __bf16 Ws[2][128 * 32];
  int tid  = threadIdx.x;
  int lane = tid & 31, wave = tid >> 5;
  int half = lane >> 4, lr = lane & 15;
  int wm = wave >> 2;        // 0..1 : 64 rows each
  int wn = wave & 3;         // 0..3 : 32 cols each
  int bm = blockIdx.y * 128, bn = blockIdx.x * 128;

  v8f acc[4][2];
#pragma unroll
  for (int mi = 0; mi < 4; ++mi)
#pragma unroll
    for (int ni = 0; ni < 2; ++ni) acc[mi][ni] = vzero8();

  auto stage = [&](int buf, int k0) {
    int q = tid;
#pragma unroll
    for (int it = 0; it < 2; ++it, q += 256) {
      int row = q >> 2, c4 = q & 3;
      const __bf16* ga = A + (size_t)(bm + row) * K + k0 + c4 * 8;
      const __bf16* gw = W + (size_t)(bn + row) * K + k0 + c4 * 8;
#if USE_ASYNC
      async_b128(ga, &As[buf][q * 8]);
      async_b128(gw, &Ws[buf][q * 8]);
#else
      ((uint4*)As[buf])[q] = *(const uint4*)ga;
      ((uint4*)Ws[buf])[q] = *(const uint4*)gw;
#endif
    }
  };

  int nk = K >> 5;
  stage(0, 0);
  for (int kt = 0; kt < nk; ++kt) {
    int buf = kt & 1;
    wait_async_zero();
    __syncthreads();                 // publish buf, and retire reads of buf^1
    if (kt + 1 < nk) stage(buf ^ 1, (kt + 1) << 5);   // overlaps with compute

    AF aF[4];
#pragma unroll
    for (int mi = 0; mi < 4; ++mi) {
      const __bf16* p = As[buf] + (wm * 64 + mi * 16 + lr) * 32 + half * 8;
      aF[mi].q[0] = *(const uint4*)p;
      aF[mi].q[1] = *(const uint4*)(p + 16);
    }
    AF bF[2];
#pragma unroll
    for (int ni = 0; ni < 2; ++ni) {
      const __bf16* p = Ws[buf] + (wn * 32 + ni * 16 + lr) * 32 + half * 8;
      bF[ni].q[0] = *(const uint4*)p;
      bF[ni].q[1] = *(const uint4*)(p + 16);
    }
#pragma unroll
    for (int mi = 0; mi < 4; ++mi)
#pragma unroll
      for (int ni = 0; ni < 2; ++ni)
        acc[mi][ni] = wmma_bf16(aF[mi].v, bF[ni].v, acc[mi][ni]);
  }

#pragma unroll
  for (int mi = 0; mi < 4; ++mi)
#pragma unroll
    for (int ni = 0; ni < 2; ++ni) {
      int col = bn + wn * 32 + ni * 16 + lr;
#pragma unroll
      for (int r = 0; r < 8; ++r) {
        int row = bm + wm * 64 + mi * 16 + r + 8 * half;
        size_t idx = (size_t)row * N + col;
        float v = acc[mi][ni][r];
        if (MODE == 0) {
          OF[idx] = v;
        } else if (MODE == 1) {
          OF[idx] += v;
        } else {
          OB[idx] = (__bf16)(0.5f * v * (1.0f + erff(v * 0.70710678118f)));
        }
      }
    }
}

// ---- flash attention: Q/K/V bf16 [B,H,T,DH] -> O bf16 [B,T,D] -------------
__global__ __launch_bounds__(128) void attn_kernel(const __bf16* __restrict__ Qb,
                                                   const __bf16* __restrict__ Kb,
                                                   const __bf16* __restrict__ Vb,
                                                   __bf16* __restrict__ O) {
  __shared__ __attribute__((aligned(16))) __bf16 Kt[32 * 64];       // [kv][dh]
  __shared__ __attribute__((aligned(16))) __bf16 Vt[64 * 32];       // [dh][kv]
  __shared__ __attribute__((aligned(16))) __bf16 Pst[4][16 * 32];   // per-wave P stage

  int tid  = threadIdx.x;
  int lane = tid & 31;
  int wave = __builtin_amdgcn_readfirstlane(tid >> 5);  // scalar wave id
  int half = lane >> 4, lr = lane & 15;

  const int nqb = TT / 64;                  // 32 query blocks
  int qb = blockIdx.x % nqb;
  int bh = blockIdx.x / nqb;
  const __bf16* Qg = Qb + (size_t)bh * TT * DHH;
  const __bf16* Kg = Kb + (size_t)bh * TT * DHH;
  const __bf16* Vg = Vb + (size_t)bh * TT * DHH;

  int q0 = qb * 64 + wave * 16;             // absolute first query row for this wave

  AF aq[2];
#pragma unroll
  for (int c = 0; c < 2; ++c) {
    const __bf16* p = Qg + (size_t)(q0 + lr) * DHH + c * 32 + half * 8;
    aq[c].q[0] = *(const uint4*)p;
    aq[c].q[1] = *(const uint4*)(p + 16);
  }

  v8f oacc[4];
#pragma unroll
  for (int n = 0; n < 4; ++n) oacc[n] = vzero8();
  float mrow[8], lrow[8];
#pragma unroll
  for (int r = 0; r < 8; ++r) { mrow[r] = -1e30f; lrow[r] = 0.0f; }

  int nkb_wg = 2 * qb + 2;                  // kv blocks the workgroup must stage
  int nkb_w  = (q0 >> 5) + 1;               // kv blocks this wave actually needs (scalar)

  for (int kb = 0; kb < nkb_wg; ++kb) {
    int kvs = kb * 32;
    // V: b128 global loads, bf16 scatter into transposed LDS tile [dh][kv]
    {
      int qd = tid;
#pragma unroll
      for (int it = 0; it < 2; ++it, qd += 128) {
        int kv = qd >> 3, dh0 = (qd & 7) * 8;
        QU u;
        u.u = *(const uint4*)(Vg + (size_t)(kvs + kv) * DHH + dh0);
#pragma unroll
        for (int i = 0; i < 8; ++i) Vt[(dh0 + i) * 32 + kv] = u.e[i];
      }
    }
    // K: contiguous b128 copy (async->LDS when available)
    {
      int qd = tid;
#pragma unroll
      for (int it = 0; it < 2; ++it, qd += 128) {
        const __bf16* g = Kg + (size_t)kvs * DHH + qd * 8;
#if USE_ASYNC
        async_b128(g, &Kt[qd * 8]);
#else
        ((uint4*)Kt)[qd] = *(const uint4*)g;
#endif
      }
    }
    wait_async_zero();
    __syncthreads();

    if (kb < nkb_w) {                       // scalar, wave-uniform branch
      v8f s[2];
      s[0] = vzero8(); s[1] = vzero8();
      // c-outer / j-inner: dependent WMMAs on s[j] are separated
#pragma unroll
      for (int c = 0; c < 2; ++c)
#pragma unroll
        for (int j = 0; j < 2; ++j) {
          AF bk;
          const __bf16* p = Kt + (j * 16 + lr) * 64 + c * 32 + half * 8;
          bk.q[0] = *(const uint4*)p;
          bk.q[1] = *(const uint4*)(p + 16);
          s[j] = wmma_bf16(aq[c].v, bk.v, s[j]);
        }

      // online softmax over this 16x32 score tile
#pragma unroll
      for (int r = 0; r < 8; ++r) {
        int grow = q0 + r + 8 * half;
        float v0 = s[0][r] * 0.125f;
        float v1 = s[1][r] * 0.125f;
        if (kvs + lr      > grow) v0 = -1e30f;
        if (kvs + 16 + lr > grow) v1 = -1e30f;
        float mx = fmaxf(v0, v1);
        mx = fmaxf(mx, __shfl_xor(mx, 1));
        mx = fmaxf(mx, __shfl_xor(mx, 2));
        mx = fmaxf(mx, __shfl_xor(mx, 4));
        mx = fmaxf(mx, __shfl_xor(mx, 8));
        float mn = fmaxf(mrow[r], mx);
        float al = expf(mrow[r] - mn);
        float p0 = expf(v0 - mn);
        float p1 = expf(v1 - mn);
        float rs = p0 + p1;
        rs += __shfl_xor(rs, 1);
        rs += __shfl_xor(rs, 2);
        rs += __shfl_xor(rs, 4);
        rs += __shfl_xor(rs, 8);
        lrow[r] = lrow[r] * al + rs;
        mrow[r] = mn;
#pragma unroll
        for (int n = 0; n < 4; ++n) oacc[n][r] = oacc[n][r] * al;
        Pst[wave][(r + 8 * half) * 32 + lr]      = (__bf16)p0;
        Pst[wave][(r + 8 * half) * 32 + 16 + lr] = (__bf16)p1;
      }

      AF pa;
      const __bf16* pp = &Pst[wave][lr * 32 + half * 8];
      pa.q[0] = *(const uint4*)pp;
      pa.q[1] = *(const uint4*)(pp + 16);
#pragma unroll
      for (int n = 0; n < 4; ++n) {
        AF bv;
        const __bf16* vp = Vt + (n * 16 + lr) * 32 + half * 8;
        bv.q[0] = *(const uint4*)vp;
        bv.q[1] = *(const uint4*)(vp + 16);
        oacc[n] = wmma_bf16(pa.v, bv.v, oacc[n]);
      }
    }
    __syncthreads();                        // retire reads before restaging
  }

  int b = bh >> 3, h = bh & 7;
#pragma unroll
  for (int r = 0; r < 8; ++r) {
    float inv = 1.0f / lrow[r];
    int t = q0 + r + 8 * half;
    size_t rowoff = ((size_t)b * TT + t) * DD + h * DHH;
#pragma unroll
    for (int n = 0; n < 4; ++n)
      O[rowoff + n * 16 + lr] = (__bf16)(oacc[n][r] * inv);
  }
}

// ---- driver ---------------------------------------------------------------
extern "C" void kernel_launch(void* const* d_in, const int* in_sizes, int n_in,
                              void* d_out, int out_size, void* d_ws, size_t ws_size,
                              hipStream_t stream) {
  const int*   ids   = (const int*)  d_in[0];
  const float* tok   = (const float*)d_in[1];
  const float* pos   = (const float*)d_in[2];
  const float* ln1w  = (const float*)d_in[3];
  const float* ln1b  = (const float*)d_in[4];
  const float* qkvw  = (const float*)d_in[5];
  const float* outw  = (const float*)d_in[6];
  const float* ln2w  = (const float*)d_in[7];
  const float* ln2b  = (const float*)d_in[8];
  const float* fc1w  = (const float*)d_in[9];
  const float* fc2w  = (const float*)d_in[10];
  const float* lnfw  = (const float*)d_in[11];
  const float* lnfb  = (const float*)d_in[12];
  const float* headw = (const float*)d_in[13];
  float* logits = (float*)d_out;

  // workspace carve-up (256B aligned)
  char* w = (char*)d_ws;
  size_t off = 0;
  auto carve = [&](size_t bytes) -> char* {
    char* p = w + off;
    off = (off + bytes + 255) & ~(size_t)255;
    return p;
  };
  float*  x     = (float*) carve((size_t)MM * DD * 4);
  float*  qkvf  = (float*) carve((size_t)MM * 3 * DD * 4);
  __bf16* Qb    = (__bf16*)carve((size_t)BB * HH * TT * DHH * 2);
  __bf16* Kb    = (__bf16*)carve((size_t)BB * HH * TT * DHH * 2);
  __bf16* Vb    = (__bf16*)carve((size_t)BB * HH * TT * DHH * 2);
  __bf16* attnO = (__bf16*)carve((size_t)MM * DD * 2);
  __bf16* actA  = (__bf16*)carve((size_t)MM * DD * 2);
  __bf16* actB  = (__bf16*)carve((size_t)MM * FFF * 2);
  __bf16* wq    = (__bf16*)carve((size_t)LL * 3 * DD * DD * 2);
  __bf16* wo    = (__bf16*)carve((size_t)LL * DD * DD * 2);
  __bf16* wf1   = (__bf16*)carve((size_t)LL * FFF * DD * 2);
  __bf16* wf2   = (__bf16*)carve((size_t)LL * DD * FFF * 2);
  __bf16* wh    = (__bf16*)carve((size_t)VV * DD * 2);

  auto cvt = [&](const float* src, __bf16* dst, size_t n) {
    cvt_kernel<<<(unsigned)((n + 255) / 256), 256, 0, stream>>>(src, dst, (int)n);
  };
  cvt(qkvw,  wq,  (size_t)LL * 3 * DD * DD);
  cvt(outw,  wo,  (size_t)LL * DD * DD);
  cvt(fc1w,  wf1, (size_t)LL * FFF * DD);
  cvt(fc2w,  wf2, (size_t)LL * DD * FFF);
  cvt(headw, wh,  (size_t)VV * DD);

  embed_kernel<<<MM, 256, 0, stream>>>(ids, tok, pos, x);

  for (int l = 0; l < LL; ++l) {
    // ---- attention ----
    ln_kernel<<<MM, 256, 0, stream>>>(x, ln1w + l * DD, ln1b + l * DD, actA);
    gemm_kernel<0><<<dim3(3 * DD / 128, MM / 128), 256, 0, stream>>>(
        actA, wq + (size_t)l * 3 * DD * DD, qkvf, nullptr, DD, 3 * DD);
    qkv_split_kernel<<<MM, 256, 0, stream>>>(qkvf, Qb, Kb, Vb);
    attn_kernel<<<BB * HH * (TT / 64), 128, 0, stream>>>(Qb, Kb, Vb, attnO);
    gemm_kernel<1><<<dim3(DD / 128, MM / 128), 256, 0, stream>>>(
        attnO, wo + (size_t)l * DD * DD, x, nullptr, DD, DD);
    // ---- MLP ----
    ln_kernel<<<MM, 256, 0, stream>>>(x, ln2w + l * DD, ln2b + l * DD, actA);
    gemm_kernel<2><<<dim3(FFF / 128, MM / 128), 256, 0, stream>>>(
        actA, wf1 + (size_t)l * FFF * DD, nullptr, actB, DD, FFF);
    gemm_kernel<1><<<dim3(DD / 128, MM / 128), 256, 0, stream>>>(
        actB, wf2 + (size_t)l * DD * FFF, x, nullptr, FFF, DD);
  }

  ln_kernel<<<MM, 256, 0, stream>>>(x, lnfw, lnfb, actA);
  gemm_kernel<0><<<dim3(VV / 128, MM / 128), 256, 0, stream>>>(
      actA, wh, logits, nullptr, DD, VV);
}
